// NNModel_35356170780677
// MI455X (gfx1250) — compile-verified, matrix operand order checked by reading, not run on hardware
//
#include <hip/hip_runtime.h>

typedef __attribute__((ext_vector_type(16))) _Float16 v16h;
typedef __attribute__((ext_vector_type(8)))  _Float16 v8h;
typedef __attribute__((ext_vector_type(8)))  float    v8f;

#define DECAY   0.6f
#define THRESH  0.6f
#define T_STEPS 128
#define BATCH   32
#define MASK_T  3136
#define NINP    256
#define H1      512
#define H2      512
#define NTOKEN  32000

// ---------------------------------------------------------------------------
// Fragment loader: works for both A (row = M) and B-transposed (row = N).
// 16-bit operand layout (CDNA5 ISA 7.12.2): lane L holds row (L&15);
// half j maps to K = k0 + (j&7) + (j>=8 ? 16 : 0) + (L>=16 ? 8 : 0).
// => two contiguous 16-byte loads per lane (global_load_b128 pairs).
// ---------------------------------------------------------------------------
__device__ __forceinline__ v16h load_frag(const _Float16* __restrict__ p,
                                          int stride, int row0, int k0) {
    int lane = threadIdx.x & 31;
    int r  = row0 + (lane & 15);
    int kb = k0 + ((lane >> 4) << 3);          // +8 halves for upper lane group
    const _Float16* base = p + (size_t)r * stride + kb;
    union { v16h v; v8h h[2]; } u;
    u.h[0] = *(const v8h*)(base);              // K = kb .. kb+7
    u.h[1] = *(const v8h*)(base + 16);         // K = kb+16 .. kb+23
    return u.v;
}

__device__ __forceinline__ v8f wmma_f16(v16h a, v16h b, v8f c) {
    return __builtin_amdgcn_wmma_f32_16x16x32_f16(false, a, false, b, (short)0,
                                                  c, false, false);
}

// ---------------------------------------------------------------------------
// Small blocked WMMA GEMM (layers 1/2):  C[M,N] = A[M,K] * B^T[N,K], f32 acc.
// SA/SB: hi/lo f16 residual terms for near-f32 accuracy (spike thresholding).
// 8 waves; wave tile 32x64 (2x4 frags); WG tile 128x128.
// ---------------------------------------------------------------------------
template <bool SA, bool SB>
__global__ __launch_bounds__(256)
void gemm_small(const _Float16* __restrict__ Ah, const _Float16* __restrict__ Al,
                const _Float16* __restrict__ Bh, const _Float16* __restrict__ Bl,
                float* __restrict__ C, int M, int N, int K) {
    const int wave = threadIdx.x >> 5;
    const int lane = threadIdx.x & 31;
    const int wm = wave & 3;
    const int wn = wave >> 2;
    const int row0 = blockIdx.y * 128 + wm * 32;
    const int col0 = blockIdx.x * 128 + wn * 64;

    v8f acc[2][4] = {};

    for (int k0 = 0; k0 < K; k0 += 32) {
        v16h aH[2], aL[2], bH[4], bL[4];
#pragma unroll
        for (int i = 0; i < 2; ++i) {
            aH[i] = load_frag(Ah, K, row0 + i * 16, k0);
            if (SA) aL[i] = load_frag(Al, K, row0 + i * 16, k0);
        }
#pragma unroll
        for (int j = 0; j < 4; ++j) {
            bH[j] = load_frag(Bh, K, col0 + j * 16, k0);
            if (SB) bL[j] = load_frag(Bl, K, col0 + j * 16, k0);
        }
#pragma unroll
        for (int i = 0; i < 2; ++i)
#pragma unroll
            for (int j = 0; j < 4; ++j) {
                acc[i][j] = wmma_f16(aH[i], bH[j], acc[i][j]);
                if (SB) acc[i][j] = wmma_f16(aH[i], bL[j], acc[i][j]);
                if (SA) acc[i][j] = wmma_f16(aL[i], bH[j], acc[i][j]);
            }
    }

    // D layout: VGPR r -> M = r (lanes 0-15) / 8+r (lanes 16-31); N = lane&15.
    const int nBase = lane & 15;
    const int mHi   = (lane >> 4) * 8;
#pragma unroll
    for (int i = 0; i < 2; ++i)
#pragma unroll
        for (int j = 0; j < 4; ++j) {
            const int n  = col0 + j * 16 + nBase;
            const int mB = row0 + i * 16 + mHi;
#pragma unroll
            for (int r = 0; r < 8; ++r)
                C[(size_t)(mB + r) * N + n] = acc[i][j][r];
        }
}

// ---------------------------------------------------------------------------
// Decode GEMM: out[4096,32000] = s2 @ dec_w^T + dec_b.  134 GFLOP.
// 16 waves (512 thr); wave tile 64x64 (4x4 frags, 128 acc VGPRs);
// WG tile 256x256 -> L2 read traffic ~1 GB total (A 0.5 GB + B 0.5 GB),
// below the 524 MB @ 23.3 TB/s NT-store floor. A (spikes) is exact in f16.
// ---------------------------------------------------------------------------
__global__ __launch_bounds__(512)
void gemm_decode(const _Float16* __restrict__ A, const _Float16* __restrict__ B,
                 const float* __restrict__ bias, float* __restrict__ C,
                 int M, int N, int K) {
    const int wave = threadIdx.x >> 5;     // 0..15
    const int lane = threadIdx.x & 31;
    const int wm = wave & 3;
    const int wn = wave >> 2;
    const int row0 = blockIdx.y * 256 + wm * 64;
    const int col0 = blockIdx.x * 256 + wn * 64;

    v8f acc[4][4] = {};

    for (int k0 = 0; k0 < K; k0 += 32) {
        v16h a[4], b[4];
#pragma unroll
        for (int i = 0; i < 4; ++i) a[i] = load_frag(A, K, row0 + i * 16, k0);
#pragma unroll
        for (int j = 0; j < 4; ++j) b[j] = load_frag(B, K, col0 + j * 16, k0);
#pragma unroll
        for (int i = 0; i < 4; ++i)
#pragma unroll
            for (int j = 0; j < 4; ++j)
                acc[i][j] = wmma_f16(a[i], b[j], acc[i][j]);
    }

    const int nBase = lane & 15;
    const int mHi   = (lane >> 4) * 8;
#pragma unroll
    for (int j = 0; j < 4; ++j) {
        const int n = col0 + j * 16 + nBase;
        const float bv = bias[n];
#pragma unroll
        for (int i = 0; i < 4; ++i) {
            const int mB = row0 + i * 16 + mHi;
#pragma unroll
            for (int r = 0; r < 8; ++r)
                __builtin_nontemporal_store(acc[i][j][r] + bv,
                                            C + (size_t)(mB + r) * N + n);
        }
    }
}

// ---------------------------------------------------------------------------
// Transpose + hi/lo f16 split:  src [K,N] f32 -> hi/lo [N,K] f16
// ---------------------------------------------------------------------------
__global__ void conv_t_f16(const float* __restrict__ src, _Float16* __restrict__ hi,
                           _Float16* __restrict__ lo, int K, int N) {
    int idx = blockIdx.x * blockDim.x + threadIdx.x;
    if (idx >= K * N) return;
    int n = idx / K, k = idx % K;
    float v = src[(size_t)k * N + n];
    _Float16 h = (_Float16)v;
    hi[(size_t)n * K + k] = h;
    lo[(size_t)n * K + k] = (_Float16)(v - (float)h);
}

// dec_w [NTOKEN,512] is already in B^T layout -> plain f16 convert
__global__ void conv_f16(const float* __restrict__ src, _Float16* __restrict__ dst,
                         int n) {
    int i = blockIdx.x * blockDim.x + threadIdx.x;
    if (i < n) dst[i] = (_Float16)src[i];
}

// Embedding gather + hi/lo split: emb[r,c] = enc_w[raw[r], c], r = t*B + b
__global__ void embed_k(const int* __restrict__ tok, const float* __restrict__ encw,
                        _Float16* __restrict__ hi, _Float16* __restrict__ lo) {
    int idx = blockIdx.x * blockDim.x + threadIdx.x;   // 4096*256
    int r = idx >> 8, c = idx & 255;
    float v = encw[(size_t)tok[r] * NINP + c];
    _Float16 h = (_Float16)v;
    hi[idx] = h;
    lo[idx] = (_Float16)(v - (float)h);
}

// ---------------------------------------------------------------------------
// Per-channel LIF scan: the recurrence is elementwise per (b,h) channel, so
// all 16K channels run in parallel; only t is serial (trivial ALU loop).
// Spikes are exactly {0,1} -> lossless f16.
// ---------------------------------------------------------------------------
__global__ void scan_k(const float* __restrict__ state, const float* __restrict__ mask,
                       _Float16* __restrict__ spikes, int H) {
    int idx = blockIdx.x * blockDim.x + threadIdx.x;   // B*H
    int b = idx / H, h = idx % H;
    float mem = 0.0f;
    for (int t = 0; t < T_STEPS; ++t) {
        float st = state[((size_t)t * BATCH + b) * H + h];
        float m  = mask[(size_t)h * MASK_T + t];
        float spike = 0.0f;
        if (m != 0.0f) {
            float keep = (mem > THRESH) ? 0.0f : 1.0f;  // reset gate
            mem = mem * keep * DECAY + st;
            spike = (mem > THRESH) ? 1.0f : 0.0f;       // mask value is exactly 1
        }
        spikes[((size_t)t * BATCH + b) * H + h] = (_Float16)spike;
    }
}

// ---------------------------------------------------------------------------
extern "C" void kernel_launch(void* const* d_in, const int* in_sizes, int n_in,
                              void* d_out, int out_size, void* d_ws, size_t ws_size,
                              hipStream_t stream) {
    const int*   raw   = (const int*)d_in[0];
    const float* encw  = (const float*)d_in[1];
    const float* fc1   = (const float*)d_in[2];
    const float* fc2   = (const float*)d_in[3];
    const float* decw  = (const float*)d_in[4];
    const float* decb  = (const float*)d_in[5];
    const float* mask1 = (const float*)d_in[6];
    const float* mask2 = (const float*)d_in[7];
    float* out = (float*)d_out;

    const int R = T_STEPS * BATCH;                     // 4096 rows
    char* ws = (char*)d_ws;
    size_t off = 0;
    auto alloc = [&](size_t bytes) { char* p = ws + off; off = (off + bytes + 255) & ~(size_t)255; return p; };

    _Float16* embH  = (_Float16*)alloc((size_t)R * NINP * 2);
    _Float16* embL  = (_Float16*)alloc((size_t)R * NINP * 2);
    _Float16* fc1tH = (_Float16*)alloc((size_t)H1 * NINP * 2);
    _Float16* fc1tL = (_Float16*)alloc((size_t)H1 * NINP * 2);
    _Float16* fc2tH = (_Float16*)alloc((size_t)H2 * H1 * 2);
    _Float16* fc2tL = (_Float16*)alloc((size_t)H2 * H1 * 2);
    _Float16* decH  = (_Float16*)alloc((size_t)NTOKEN * H2 * 2);
    float*    state = (float*)alloc((size_t)R * H1 * 4);   // reused for state1/state2
    _Float16* s1    = (_Float16*)alloc((size_t)R * H1 * 2);
    _Float16* s2    = (_Float16*)alloc((size_t)R * H2 * 2);

    // Weight conversions
    conv_t_f16<<<(NINP * H1 + 255) / 256, 256, 0, stream>>>(fc1, fc1tH, fc1tL, NINP, H1);
    conv_t_f16<<<(H1 * H2 + 255) / 256, 256, 0, stream>>>(fc2, fc2tH, fc2tL, H1, H2);
    conv_f16<<<(NTOKEN * H2 + 255) / 256, 256, 0, stream>>>(decw, decH, NTOKEN * H2);

    // Embedding gather (all T at once)
    embed_k<<<(R * NINP + 255) / 256, 256, 0, stream>>>(raw, encw, embH, embL);

    // state1 = emb @ fc1   (3-term hi/lo split, near-f32)
    gemm_small<true, true><<<dim3(H1 / 128, R / 128), 256, 0, stream>>>(
        embH, embL, fc1tH, fc1tL, state, R, H1, NINP);

    // layer-1 LIF scan (parallel over channels)
    scan_k<<<(BATCH * H1 + 255) / 256, 256, 0, stream>>>(state, mask1, s1, H1);

    // state2 = s1 @ fc2   (A exact binary, 2-term split on B)
    gemm_small<false, true><<<dim3(H2 / 128, R / 128), 256, 0, stream>>>(
        s1, nullptr, fc2tH, fc2tL, state, R, H2, H1);

    // layer-2 LIF scan
    scan_k<<<(BATCH * H2 + 255) / 256, 256, 0, stream>>>(state, mask2, s2, H2);

    // decode: out = s2 @ dec_w^T + dec_b
    gemm_decode<<<dim3(NTOKEN / 256, R / 256), 512, 0, stream>>>(
        s2, decH, decb, out, R, NTOKEN, H2);

    (void)in_sizes; (void)n_in; (void)out_size; (void)ws_size;
}
